// FlowBasedDensityOverflow_57397942944087
// MI455X (gfx1250) — compile-verified, matrix operand order checked by reading, not run on hardware
//
#include <hip/hip_runtime.h>
#include <hip/hip_bf16.h>

// CDNA5 / gfx1250 flood-fill (3 centered windows) + full-grid scatter.
// - Output streaming (201MB) dominates: non-temporal b128 stores, ~8.6us @ 23.3TB/s.
// - Fill runs entirely in LDS as bitmasks (wave32 ballot pack, Kogge-Stone run smear).
// - Final popcount reduction goes through v_wmma_f32_16x16x32_f16 (ones-matmul rowsum).

typedef __attribute__((ext_vector_type(16))) _Float16 v16h;
typedef __attribute__((ext_vector_type(8)))  float    v8f;
typedef __attribute__((ext_vector_type(4)))  float    v4f;

#define GRID 4096
#define CXY  2048   // center (GRID/2, GRID/2)

// Smear set bits of x through maximal runs of 1s in f, both directions,
// within one 32-bit word (Kogge-Stone with propagate chain). A source bit
// does NOT need f at its own position (matches reference: seed propagates
// out even when the seed cell itself is blocked); every reached target
// position requires f.
__device__ __forceinline__ unsigned smear_runs(unsigned x, unsigned f) {
    unsigned y = x, p = f, q;
    y |= (y << 1)  & p;  q  = p & (p << 1);
    y |= (y << 2)  & q;  q &= (q << 2);
    y |= (y << 4)  & q;  q &= (q << 4);
    y |= (y << 8)  & q;  q &= (q << 8);
    y |= (y << 16) & q;
    unsigned z = x;
    z |= (z >> 1)  & p;  q  = p & (p >> 1);
    z |= (z >> 2)  & q;  q &= (q >> 2);
    z |= (z >> 4)  & q;  q &= (q >> 4);
    z |= (z >> 8)  & q;  q &= (q >> 8);
    z |= (z >> 16) & q;
    return y | z;
}

// Streaming zero of the whole output (masks + counts). NT hint: write-once
// data should not displace useful L2 lines.
__global__ void fbdo_zero_kernel(float* __restrict__ out, long long n) {
    long long i = ((long long)blockIdx.x * blockDim.x + threadIdx.x) * 4;
    if (i + 4 <= n) {
        v4f z = 0.0f;
        __builtin_nontemporal_store(z, (v4f*)(out + i));
    } else {
        for (; i < n; ++i) out[i] = 0.0f;
    }
}

// One workgroup per window size. blockIdx.x = 0,1,2 -> S = 128,256,512.
__global__ __launch_bounds__(1024)
void fbdo_flood_kernel(const float* __restrict__ in, float* __restrict__ out) {
    __shared__ unsigned freeb[512 * 16];   // 32 KB: free-cell bitmask
    __shared__ unsigned reach[512 * 16];   // 32 KB: reached bitmask
    __shared__ float    psum[1024];        // per-thread popcount partials
    __shared__ int      flag;

    const int widx  = blockIdx.x;
    const int lg    = 7 + widx;            // log2(S): 7,8,9
    const int S     = 1 << lg;             // 128,256,512
    const int WORDS = S >> 5;              // dwords per row: 4,8,16
    const int half  = S >> 1;
    const int sx    = CXY - half;          // window origin (row)
    const int sy    = CXY - half;          // window origin (col)
    const int tid   = threadIdx.x;
    const int lane  = tid & 31;
    const int wave  = tid >> 5;

    // ---- init reach = 0 ----
    for (int i = tid; i < S * WORDS; i += (int)blockDim.x) reach[i] = 0u;

    // ---- load window, pack free bits with wave32 ballot ----
    const int nwords = (S * S) >> 5;
    const int nwaves = (int)blockDim.x >> 5;   // 32
    for (int i = wave; i < nwords; i += nwaves) {
        int cellBase = i << 5;
        int r = cellBase >> lg;
        int cb = cellBase & (S - 1);
        const float* p = in + (size_t)(sx + r) * GRID + (size_t)(sy + cb + lane);
        __builtin_prefetch(p + GRID, 0, 0);    // global_prefetch_b8 (next rows)
        float v = *p;
        unsigned bits = __builtin_amdgcn_ballot_w32(v == 0.0f);
        if (lane == 0) freeb[r * WORDS + (cb >> 5)] = bits;
    }
    __syncthreads();

    // ---- seed (always tagged, even if blocked) ----
    const int      seedw   = half >> 5;
    const unsigned seedbit = 1u << (half & 31);
    if (tid == 0) reach[half * WORDS + seedw] = seedbit;

    // ---- thread tile: 8-row strip x one dword column ----
    const int  w       = tid & (WORDS - 1);
    const int  sid     = tid >> (lg - 5);
    const int  nstrips = S >> 3;
    const bool active  = (sid < nstrips);
    const int  r0      = sid << 3;

    unsigned f[8], sm[8];
    unsigned t[8] = {0, 0, 0, 0, 0, 0, 0, 0};
    if (active) {
        for (int i = 0; i < 8; ++i) {
            f[i]  = freeb[(r0 + i) * WORDS + w];
            sm[i] = (((r0 + i) == half) && (w == seedw)) ? seedbit : 0u;
        }
    }
    __syncthreads();

    // ---- converge: monotone chaotic relaxation to the reachable set ----
    for (;;) {
        if (tid == 0) flag = 0;
        __syncthreads();

        int changed = 0;
        if (active) {
            unsigned o[8], inj[8];
            for (int i = 0; i < 8; ++i) { t[i] = o[i] = reach[(r0 + i) * WORDS + w]; }
            unsigned up = (r0 > 0)     ? reach[(r0 - 1) * WORDS + w] : 0u;
            unsigned dn = (r0 + 8 < S) ? reach[(r0 + 8) * WORDS + w] : 0u;
            for (int i = 0; i < 8; ++i) {
                unsigned l = (w > 0)         ? ((reach[(r0 + i) * WORDS + w - 1] >> 31) & 1u) : 0u;
                unsigned r = (w < WORDS - 1) ? ((reach[(r0 + i) * WORDS + w + 1] & 1u) << 31) : 0u;
                inj[i] = l | r;
            }
            // down-sweep then up-sweep: full vertical prop inside strip,
            // full horizontal prop inside word each sweep.
            unsigned prev = up;
            for (int i = 0; i < 8; ++i) {
                unsigned x = ((t[i] | prev | inj[i]) & f[i]) | sm[i];
                x = smear_runs(x, f[i]);
                t[i] = x;
                prev = x;
            }
            unsigned nxt = dn;
            for (int i = 7; i >= 0; --i) {
                unsigned x = ((t[i] | nxt | inj[i]) & f[i]) | sm[i];
                x = smear_runs(x, f[i]);
                t[i] = x;
                nxt = x;
            }
            for (int i = 0; i < 8; ++i) {
                changed |= (t[i] != o[i]);
                reach[(r0 + i) * WORDS + w] = t[i];
            }
        }
        if (changed) atomicOr(&flag, 1);
        __syncthreads();
        int done = (flag == 0);
        __syncthreads();
        if (done) break;
    }

    // ---- scatter window region of this mask (rest already zeroed) ----
    float* outMask = out + (size_t)widx * ((size_t)GRID * GRID);
    for (int idx = tid; idx < S * S; idx += (int)blockDim.x) {
        int r = idx >> lg;
        int c = idx & (S - 1);
        unsigned bits = reach[r * WORDS + (c >> 5)];
        outMask[(size_t)(sx + r) * GRID + (size_t)(sy + c)] =
            ((bits >> (c & 31)) & 1u) ? 1.0f : 0.0f;
    }

    // ---- count via WMMA ones-matmul rowsum reduction ----
    float part = 0.0f;
    if (active) {
        int pc = 0;
        for (int i = 0; i < 8; ++i) pc += __popc((int)t[i]);
        part = (float)pc;                     // <= 256, f16-exact after pairing
    }
    psum[tid] = part;
    __syncthreads();

    if (tid < 32) {                            // wave 0, full EXEC within wave
        v16h a, b;
        for (int j = 0; j < 16; ++j) {
            int k = (lane << 4) + j;           // 0..511
            a[j] = (_Float16)(psum[2 * k] + psum[2 * k + 1]);  // <= 512, exact
            b[j] = (_Float16)1.0f;
        }
        v8f c = {};
        // D[m][n] = sum_k A[m][k] * 1 : every D element is a row-sum.
        v8f d = __builtin_amdgcn_wmma_f32_16x16x32_f16(
            /*neg_a=*/false, a, /*neg_b=*/false, b,
            /*c_mod=*/(short)0, c, /*reuse_a=*/false, /*reuse_b=*/false);
        float s = d[0] + d[1] + d[2] + d[3] + d[4] + d[5] + d[6] + d[7];
        for (int off = 16; off > 0; off >>= 1) s += __shfl_xor(s, off, 32);
        // Each of the 16 row-sums appears in all 16 columns -> total = s/16.
        if (lane == 0) {
            int* cnt = (int*)(out + (size_t)3 * GRID * GRID);  // counts: int32 bits
            cnt[widx] = (int)(s * (1.0f / 16.0f) + 0.5f);
        }
    }
}

extern "C" void kernel_launch(void* const* d_in, const int* in_sizes, int n_in,
                              void* d_out, int out_size, void* d_ws, size_t ws_size,
                              hipStream_t stream) {
    (void)in_sizes; (void)n_in; (void)d_ws; (void)ws_size;
    const float* mask = (const float*)d_in[0];
    float* out = (float*)d_out;

    // 1) zero the full output (3 masks + 3 count slots), streaming NT stores
    long long n = (long long)out_size;
    long long chunks = (n + 3) / 4;
    int threads = 256;
    long long blocks = (chunks + threads - 1) / threads;
    fbdo_zero_kernel<<<(unsigned)blocks, threads, 0, stream>>>(out, n);

    // 2) one workgroup per window size: LDS bitmask flood fill + scatter + count
    fbdo_flood_kernel<<<3, 1024, 0, stream>>>(mask, out);
}
